// SaliencyFuse_85907935855173
// MI455X (gfx1250) — matrix operand order repaired and checked by reading, not compiled, not run on hardware
//
#include <hip/hip_runtime.h>
#include <hip/hip_bf16.h>
#include <math.h>

#define DEVFN static __device__ __forceinline__

typedef __attribute__((ext_vector_type(16))) _Float16 v16h;
typedef __attribute__((ext_vector_type(8)))  _Float16 v8h;
typedef __attribute__((ext_vector_type(8)))  float    v8f;

#define NB 4
#define H0 512
#define W0 640
#define HW (H0 * W0)

// ---------------- helpers ----------------

// order-preserving float <-> uint key (deterministic atomic min/max)
DEVFN unsigned fkey(float f) {
    unsigned u = __float_as_uint(f);
    return (u & 0x80000000u) ? ~u : (u | 0x80000000u);
}
DEVFN float funkey(unsigned k) {
    unsigned v = (k & 0x80000000u) ? (k & 0x7fffffffu) : ~k;
    return __uint_as_float(v);
}

DEVFN int refl(int i, int n) { if (i < 0) i = -i; if (i >= n) i = 2 * n - 2 - i; return i; }

// jax.image.resize 'bilinear' sampling of plane p (h x w) at output coord (y,x) of (hout x wout)
DEVFN float bl_sample(const float* __restrict__ p, int h, int w, int hout, int wout, int y, int x) {
    float sy = (y + 0.5f) * (float)h / (float)hout - 0.5f;
    float sx = (x + 0.5f) * (float)w / (float)wout - 0.5f;
    float fy0 = floorf(sy), fx0 = floorf(sx);
    int y0 = (int)fy0, x0 = (int)fx0;
    float fy = sy - fy0, fx = sx - fx0;
    int y0c = min(max(y0, 0), h - 1), y1c = min(max(y0 + 1, 0), h - 1);
    int x0c = min(max(x0, 0), w - 1), x1c = min(max(x0 + 1, 0), w - 1);
    float v00 = p[y0c * w + x0c], v01 = p[y0c * w + x1c];
    float v10 = p[y1c * w + x0c], v11 = p[y1c * w + x1c];
    return v00 * (1.f - fy) * (1.f - fx) + v01 * (1.f - fy) * fx +
           v10 * fy * (1.f - fx) + v11 * fy * fx;
}

// block(256)-wide min/max reduce then one atomic per block
DEVFN void redMinMax(unsigned kmn, unsigned kmx, unsigned* pair) {
    __shared__ unsigned smn[256];
    __shared__ unsigned smx[256];
    int t = threadIdx.x;
    __syncthreads();
    smn[t] = kmn; smx[t] = kmx;
    __syncthreads();
    for (int s = 128; s > 0; s >>= 1) {
        if (t < s) {
            if (smn[t + s] < smn[t]) smn[t] = smn[t + s];
            if (smx[t + s] > smx[t]) smx[t] = smx[t + s];
        }
        __syncthreads();
    }
    if (t == 0) { atomicMin(&pair[0], smn[0]); atomicMax(&pair[1], smx[0]); }
}

// ---------------- kernels ----------------

__global__ void k_init_pairs(unsigned* base, int npairs) {
    int i = blockIdx.x * blockDim.x + threadIdx.x;
    if (i < npairs) { base[2 * i] = 0xFFFFFFFFu; base[2 * i + 1] = 0u; }
}

__global__ void k_zero(float* p, long n) {
    long i = (long)blockIdx.x * blockDim.x + threadIdx.x;
    long st = (long)gridDim.x * blockDim.x;
    for (; i < n; i += st) p[i] = 0.f;
}

__global__ void k_rgb_lab(const float* __restrict__ img, float* __restrict__ l_out,
                          float* __restrict__ ab_out) {
    int idx = blockIdx.x * 256 + threadIdx.x;
    if (idx >= HW) return;
    int b = blockIdx.z;
    const float* p = img + (size_t)b * 3 * HW;
    auto lin = [](float v) { return (v > 0.04045f) ? powf((v + 0.055f) / 1.055f, 2.4f) : v / 12.92f; };
    float lr = lin(p[idx]), lg = lin(p[HW + idx]), lb = lin(p[2 * HW + idx]);
    float X = (0.412453f * lr + 0.35758f * lg + 0.180423f * lb) / 0.95047f;
    float Y = (0.212671f * lr + 0.71516f * lg + 0.072169f * lb);
    float Z = (0.019334f * lr + 0.119193f * lg + 0.950227f * lb) / 1.08883f;
    auto fl = [](float t) { return (t > 0.008856f) ? cbrtf(t) : 7.787f * t + 16.f / 116.f; };
    float fx = fl(X), fy = fl(Y), fz = fl(Z);
    l_out[(size_t)b * HW + idx] = (116.f * fy - 16.f) / 100.f;
    ab_out[((size_t)b * 2 + 0) * HW + idx] = 500.f * (fx - fy);
    ab_out[((size_t)b * 2 + 1) * HW + idx] = 200.f * (fy - fz);
}

// 5x5 gaussian (zero pad) + 2x decimation
__global__ void k_pyr_down(const float* __restrict__ src, float* __restrict__ dst,
                           int hin, int win, int hout, int wout) {
    int idx = blockIdx.x * 256 + threadIdx.x;
    if (idx >= hout * wout) return;
    int b = blockIdx.z;
    int y = idx / wout, x = idx % wout;
    const float gw[5] = {0.054488685f, 0.244201342f, 0.402619947f, 0.244201342f, 0.054488685f};
    const float* sp = src + (size_t)b * hin * win;
    float s = 0.f;
#pragma unroll
    for (int i = 0; i < 5; ++i) {
        int iy = 2 * y + i - 2;
        if (iy < 0 || iy >= hin) continue;
#pragma unroll
        for (int j = 0; j < 5; ++j) {
            int ix = 2 * x + j - 2;
            if (ix < 0 || ix >= win) continue;
            s += gw[i] * gw[j] * sp[iy * win + ix];
        }
    }
    dst[(size_t)b * hout * wout + idx] = s;
}

// intensity: global min/max of |hi - up(lo)| over the hi-level domain (all batches -> one pair)
__global__ void k_int_stats(const float* __restrict__ hi, const float* __restrict__ lo,
                            int hh, int wh, int hl, int wl, unsigned* pair) {
    int idx = blockIdx.x * 256 + threadIdx.x;
    int b = blockIdx.z;
    unsigned kmn = 0xFFFFFFFFu, kmx = 0u;
    if (idx < hh * wh) {
        int y = idx / wh, x = idx % wh;
        const float* hp = hi + (size_t)b * hh * wh;
        const float* lp = lo + (size_t)b * hl * wl;
        float v = fabsf(hp[idx] - bl_sample(lp, hl, wl, hh, wh, y, x));
        unsigned k = fkey(v); kmn = k; kmx = k;
    }
    redMinMax(kmn, kmx, pair);
}

// tot += nn_upsample(mm_global(|hi - up(lo)|))   (recompute diff, normalize pointwise)
__global__ void k_int_accum(const float* __restrict__ hi, const float* __restrict__ lo,
                            int hh, int wh, int hl, int wl, int sh,
                            const unsigned* pair, float* __restrict__ tot) {
    int idx = blockIdx.x * 256 + threadIdx.x;
    if (idx >= HW) return;
    int b = blockIdx.z;
    int y = idx / W0, x = idx % W0;
    int hy = y >> sh, hx = x >> sh;
    const float* hp = hi + (size_t)b * hh * wh;
    const float* lp = lo + (size_t)b * hl * wl;
    float v = fabsf(hp[hy * wh + hx] - bl_sample(lp, hl, wl, hh, wh, hy, hx));
    float mn = funkey(pair[0]), mx = funkey(pair[1]);
    tot[(size_t)b * HW + idx] += (v - mn) / (mx - mn + 1e-6f);
}

__global__ void k_minmax(const float* __restrict__ p, long n, unsigned* pair) {
    long i = (long)blockIdx.x * 256 + threadIdx.x;
    long st = (long)gridDim.x * 256;
    unsigned kmn = 0xFFFFFFFFu, kmx = 0u;
    for (; i < n; i += st) {
        unsigned k = fkey(p[i]);
        if (k < kmn) kmn = k;
        if (k > kmx) kmx = k;
    }
    redMinMax(kmn, kmx, pair);
}

__global__ void k_minmax_planes(const float* __restrict__ p, int n, unsigned* pairs) {
    int pl = blockIdx.z;
    const float* q = p + (size_t)pl * n;
    unsigned kmn = 0xFFFFFFFFu, kmx = 0u;
    for (int i = blockIdx.x * 256 + threadIdx.x; i < n; i += gridDim.x * 256) {
        unsigned k = fkey(q[i]);
        if (k < kmn) kmn = k;
        if (k > kmx) kmx = k;
    }
    redMinMax(kmn, kmx, &pairs[2 * pl]);
}

__global__ void k_apply_mm(const float* __restrict__ in, float* __restrict__ out, long n,
                           const unsigned* pair) {
    long i = (long)blockIdx.x * 256 + threadIdx.x;
    if (i >= n) return;
    float mn = funkey(pair[0]), mx = funkey(pair[1]);
    out[i] = (in[i] - mn) / (mx - mn + 1e-6f);
}

// ---- Gabor bank via WMMA GEMM ----
// D(16 px x 16 orient) = A(16 px x 96 taps, im2col) x B(96 taps x 16 orient)
// 16 orientation columns = bank0(8) ++ bank1(8), so each pyramid level is convolved once.
// Tile staging uses CDNA5 async memory->LDS loads (ASYNCcnt), bypassing VGPRs.
__global__ __launch_bounds__(256) void k_gabor16(const float* __restrict__ src,
                                                 const float* __restrict__ gab2, // 16*81 f32
                                                 float* __restrict__ out,        // NB*16*hc*wc
                                                 int hc, int wc) {
    __shared__ __align__(16) _Float16 Wls[16][96];    // [orient][tap], zero-padded taps 81..95
    __shared__ __align__(16) float    stageF[16 * 24]; // raw f32 halo tile (async target)
    __shared__ __align__(16) _Float16 tileF[16 * 32];  // f16 halo tile, 16 x 24 used, stride 32
    const int tid = threadIdx.x;
    const int lane = tid & 31;
    const int wave = tid >> 5; // 0..7 -> output row within tile
    const int b = blockIdx.z;
    const int tx0 = blockIdx.x * 16;
    const int ty0 = blockIdx.y * 8;

    // weights, orientation-major, f16
    for (int i = tid; i < 16 * 96; i += 256) {
        int m = i / 96, k = i % 96;
        Wls[m][k] = (_Float16)((k < 81) ? gab2[m * 81 + k] : 0.0f);
    }

    // ---- async global -> LDS staging of the f32 halo tile (clamped addresses; OOB
    //      zeros are applied during the f16 conversion pass, so no LDS write races) ----
    const float* sp = src + (size_t)b * hc * wc;
    for (int i = tid; i < 16 * 24; i += 256) {
        int r = i / 24, cL = i % 24;
        int gy = min(max(ty0 - 4 + r, 0), hc - 1);
        int gx = min(max(tx0 - 4 + cL, 0), wc - 1);
        const float* gp = sp + (size_t)gy * wc + gx;
        unsigned loff = (unsigned)(unsigned long long)(&stageF[i]); // LDS byte offset
        asm volatile("global_load_async_to_lds_b32 %0, %1, off"
                     :
                     : "v"(loff), "v"(gp)
                     : "memory");
    }
    asm volatile("s_wait_asynccnt 0x0" ::: "memory");
    __syncthreads(); // all waves drained their async writes -> stageF complete

    // f32 -> f16 conversion with zero padding outside the plane
    for (int i = tid; i < 16 * 24; i += 256) {
        int r = i / 24, cL = i % 24;
        int gy = ty0 - 4 + r, gx = tx0 - 4 + cL;
        bool inb = (gy >= 0) && (gy < hc) && (gx >= 0) && (gx < wc);
        tileF[r * 32 + cL] = (_Float16)(inb ? stageF[i] : 0.0f);
    }
    __syncthreads();

    const int ln = lane & 15;
    const int hi = lane >> 4;
    const int pbase = wave * 32 + ln; // this lane's patch origin in tileF (pixel = A-row)
    v8f acc = {};
#pragma unroll
    for (int kk = 0; kk < 96; kk += 32) {
        // A fragment (16-bit A 16x32 layout): lane row m = ln; element e -> tap:
        //   e<8 : k = kk + hi*8 + e ; e>=8 : k = kk + 16 + hi*8 + (e-8)
        // All tap->(dy,dx) maps are compile-time; hi only selects between two constant
        // LDS offsets (one v_cndmask), no divergence, no runtime division.
        v16h a;
#pragma unroll
        for (int e = 0; e < 16; ++e) {
            const int k0 = (e < 8) ? (kk + e) : (kk + 16 + (e - 8)); // hi==0 tap
            const int k1 = k0 + 8;                                   // hi==1 tap
            const bool ok0 = (k0 < 81), ok1 = (k1 < 81);
            const int c0 = (k0 / 9) * 32 + (k0 % 9);
            const int c1 = (k1 / 9) * 32 + (k1 % 9);
            _Float16 r;
            if (ok0 && ok1) {
                r = tileF[pbase + (hi ? c1 : c0)];
            } else if (ok0) {
                _Float16 t = tileF[pbase + c0];
                r = hi ? (_Float16)0.f : t;
            } else if (ok1) {
                _Float16 t = tileF[pbase + c1];
                r = hi ? t : (_Float16)0.f;
            } else {
                r = (_Float16)0.f;
            }
            a[e] = r;
        }
        // B fragment: lane col n = ln; e<8 : k = kk + hi*8 + e ; e>=8 : k = kk + 16 + hi*8 + (e-8)
        // -> two aligned 16B LDS loads + register concat.
        v8h b0 = *reinterpret_cast<const v8h*>(&Wls[ln][kk + hi * 8]);
        v8h b1 = *reinterpret_cast<const v8h*>(&Wls[ln][kk + 16 + hi * 8]);
        v16h bm = __builtin_shufflevector(b0, b1, 0, 1, 2, 3, 4, 5, 6, 7,
                                          8, 9, 10, 11, 12, 13, 14, 15);
        acc = __builtin_amdgcn_wmma_f32_16x16x32_f16(false, a, false, bm,
                                                     (short)0, acc, false, false);
    }
    // D layout: lane holds orientation n = ln, pixels m = r + 8*hi  (all 32 lanes useful)
    int gy = ty0 + wave;
    float* op = out + ((size_t)b * 16 + ln) * hc * wc;
    if (gy < hc) {
#pragma unroll
        for (int r = 0; r < 8; ++r) {
            int gx = tx0 + r + 8 * hi;
            if (gx < wc) op[(size_t)gy * wc + gx] = acc[r];
        }
    }
}

// per-(b,orient)-plane min/max of |Oh - up(Ol)| over hi-level domain; 16-channel conv buffers
__global__ void k_ori_stats(const float* __restrict__ Oh, const float* __restrict__ Ol,
                            int hh, int wh, int hl, int wl, int choff, unsigned* pairs) {
    int pl = blockIdx.z; // b*8+ch
    int b = pl >> 3, ch = pl & 7;
    int idx = blockIdx.x * 256 + threadIdx.x;
    unsigned kmn = 0xFFFFFFFFu, kmx = 0u;
    if (idx < hh * wh) {
        int y = idx / wh, x = idx % wh;
        const float* hp = Oh + ((size_t)b * 16 + ch) * hh * wh;
        const float* lp = Ol + ((size_t)b * 16 + choff + ch) * hl * wl;
        float v = fabsf(hp[idx] - bl_sample(lp, hl, wl, hh, wh, y, x));
        unsigned k = fkey(v); kmn = k; kmx = k;
    }
    redMinMax(kmn, kmx, &pairs[2 * pl]);
}

__global__ void k_ori_accum(const float* __restrict__ Oh, const float* __restrict__ Ol,
                            int hh, int wh, int hl, int wl, int sh, int choff,
                            const unsigned* pairs, float* __restrict__ tot8) {
    int idx = blockIdx.x * 256 + threadIdx.x;
    if (idx >= HW) return;
    int pl = blockIdx.z;
    int b = pl >> 3, ch = pl & 7;
    int y = idx / W0, x = idx % W0;
    int hy = y >> sh, hx = x >> sh;
    const float* hp = Oh + ((size_t)b * 16 + ch) * hh * wh;
    const float* lp = Ol + ((size_t)b * 16 + choff + ch) * hl * wl;
    float v = fabsf(hp[hy * wh + hx] - bl_sample(lp, hl, wl, hh, wh, hy, hx));
    float mn = funkey(pairs[2 * pl]), mx = funkey(pairs[2 * pl + 1]);
    tot8[(size_t)pl * HW + idx] += (v - mn) / (mx - mn + 1e-6f);
}

// s = sum_ch mm_spatial(tot8), plus global min/max of s
__global__ void k_ori_s(const float* __restrict__ tot8, const unsigned* pairs,
                        float* __restrict__ s, unsigned* pair) {
    int idx = blockIdx.x * 256 + threadIdx.x; // HW % 256 == 0, always valid
    int b = blockIdx.z;
    float acc = 0.f;
#pragma unroll
    for (int ch = 0; ch < 8; ++ch) {
        int pl = b * 8 + ch;
        float mn = funkey(pairs[2 * pl]), mx = funkey(pairs[2 * pl + 1]);
        acc += (tot8[(size_t)pl * HW + idx] - mn) / (mx - mn + 1e-6f);
    }
    s[(size_t)b * HW + idx] = acc;
    unsigned k = fkey(acc);
    redMinMax(k, k, pair);
}

// deterministic per-batch L2 distance: one block per batch
__global__ void k_dist(const float* __restrict__ a, const float* __restrict__ b,
                       int n, float* outv) {
    int bb = blockIdx.x;
    const float* ap = a + (size_t)bb * n;
    const float* bp = b + (size_t)bb * n;
    float s = 0.f;
    for (int i = threadIdx.x; i < n; i += 256) { float d = ap[i] - bp[i]; s += d * d; }
    __shared__ float sh[256];
    sh[threadIdx.x] = s;
    __syncthreads();
    for (int st = 128; st > 0; st >>= 1) {
        if (threadIdx.x < st) sh[threadIdx.x] += sh[threadIdx.x + st];
        __syncthreads();
    }
    if (threadIdx.x == 0) outv[bb] = sqrtf(sh[0] + 1e-6f);
}

__global__ void k_fusion(const float* __restrict__ iv, const float* __restrict__ ii,
                         const float* __restrict__ ov, const float* __restrict__ oi,
                         const float* __restrict__ lv, const float* __restrict__ ir,
                         const float* __restrict__ dists,
                         float* __restrict__ fv, float* __restrict__ fi,
                         unsigned* pairFV, unsigned* pairFI) {
    int idx = blockIdx.x * 256 + threadIdx.x;
    int b = blockIdx.z;
    size_t p = (size_t)b * HW + idx;
    float d_iv = dists[0 * NB + b], d_ii = dists[1 * NB + b];
    float d_ov = dists[2 * NB + b], d_oi = dists[3 * NB + b];
    float SMv = d_iv / (d_iv + d_ov) * ov[p] + d_ov / (d_iv + d_ov) * iv[p];
    float SMi = d_ii / (d_ii + d_oi) * oi[p] + d_oi / (d_ii + d_oi) * ii[p];
    float fvv = SMv / (SMv + SMi + 1e-6f) * lv[p];
    float fii = SMi / (SMv + SMi + 1e-6f) * ir[p];
    fv[p] = fvv; fi[p] = fii;
    unsigned kv = fkey(fvv), ki = fkey(fii);
    redMinMax(kv, kv, pairFV);
    redMinMax(ki, ki, pairFI);
}

__global__ void k_fuse(const float* __restrict__ fv, const float* __restrict__ fi,
                       const unsigned* pairFV, const unsigned* pairFI,
                       float* __restrict__ fused) {
    int idx = blockIdx.x * 256 + threadIdx.x;
    int b = blockIdx.z;
    size_t p = (size_t)b * HW + idx;
    float mn0 = funkey(pairFV[0]), mx0 = funkey(pairFV[1]);
    float mn1 = funkey(pairFI[0]), mx1 = funkey(pairFI[1]);
    fused[p] = (fv[p] - mn0) / (mx0 - mn0 + 1e-6f) + (fi[p] - mn1) / (mx1 - mn1 + 1e-6f);
}

__global__ void k_bilateral(const float* __restrict__ F, float* __restrict__ enh,
                            unsigned* pair) {
    int idx = blockIdx.x * 256 + threadIdx.x;
    int b = blockIdx.z;
    int y = idx / W0, x = idx % W0;
    const float* fp = F + (size_t)b * HW;
    float c = fp[y * W0 + x];
    float accw = 0.f, accv = 0.f;
    const float inv2ss = 1.f / (2.f * 1.6f * 1.6f);
#pragma unroll
    for (int dy = -2; dy <= 2; ++dy) {
        int iy = refl(y + dy, H0);
#pragma unroll
        for (int dx = -2; dx <= 2; ++dx) {
            int ix = refl(x + dx, W0);
            float pv = fp[iy * W0 + ix];
            float sw = __expf(-(float)(dy * dy + dx * dx) * inv2ss);
            float t = (pv - c) / 0.1f;
            float cw = __expf(-0.5f * t * t);
            float w = sw * cw;
            accw += w; accv += w * pv;
        }
    }
    float blur = accv / accw;
    float e = c + logf((c - blur) + 1.f) / 100.f;
    enh[(size_t)b * HW + idx] = e;
    unsigned k = fkey(e);
    redMinMax(k, k, pair);
}

__global__ void k_lab_out(const float* __restrict__ enh, const unsigned* pair,
                          const float* __restrict__ lab_ab, float* __restrict__ out) {
    int idx = blockIdx.x * 256 + threadIdx.x;
    if (idx >= HW) return;
    int b = blockIdx.z;
    float mn = funkey(pair[0]), mx = funkey(pair[1]);
    float L = (enh[(size_t)b * HW + idx] - mn) / (mx - mn + 1e-6f) * 100.f;
    float a = lab_ab[((size_t)b * 2 + 0) * HW + idx];
    float bb = lab_ab[((size_t)b * 2 + 1) * HW + idx];
    float fy = (L + 16.f) / 116.f, fx = fy + a / 500.f, fz = fy - bb / 200.f;
    float xr = fx * fx * fx, yr = fy * fy * fy, zr = fz * fz * fz;
    const float EPS = 0.008856f, K = 1.f / 7.787f, D = 16.f / 116.f;
    float X = ((xr > EPS) ? xr : (fx - D) * K) * 0.95047f;
    float Y = ((yr > EPS) ? yr : (fy - D) * K);
    float Z = ((zr > EPS) ? zr : (fz - D) * K) * 1.08883f;
    float r  =  3.2404813432f * X - 1.5371515163f * Y - 0.4985363262f * Z;
    float g  = -0.96925495f   * X + 1.8759900015f * Y + 0.0415559266f * Z;
    float bl =  0.0556466391f * X - 0.2040413384f * Y + 1.0573110696f * Z;
    auto gam = [](float v) {
        float s = (v > 0.0031308f) ? 1.055f * powf(fmaxf(v, 1e-8f), 1.f / 2.4f) - 0.055f
                                   : 12.92f * v;
        return fminf(fmaxf(s, 0.f), 1.f);
    };
    out[((size_t)b * 3 + 0) * HW + idx] = gam(r);
    out[((size_t)b * 3 + 1) * HW + idx] = gam(g);
    out[((size_t)b * 3 + 2) * HW + idx] = gam(bl);
}

// ---------------- host launch ----------------

extern "C" void kernel_launch(void* const* d_in, const int* in_sizes, int n_in,
                              void* d_out, int out_size, void* d_ws, size_t ws_size,
                              hipStream_t stream) {
    (void)in_sizes; (void)n_in; (void)out_size; (void)ws_size;
    const float* img_vis = (const float*)d_in[0];
    const float* img_ir  = (const float*)d_in[1];
    const float* gabor   = (const float*)d_in[2]; // (4,8,9,9); banks 0,1 contiguous = 16x81
    float* out = (float*)d_out;
    float* Wf = (float*)d_ws;
    unsigned* stats = (unsigned*)d_ws; // pairs: 0=scratch, 1=fv, 2=fi, 8..39=plane stats
    float* dists = Wf + 96;            // 16 floats inside the 128-float header

    static const int LHh[8] = {512, 256, 128, 64, 32, 16, 8, 4};
    static const int LWw[8] = {640, 320, 160, 80, 40, 20, 10, 5};

    size_t off = 128;
    auto alloc = [&](size_t n) { size_t o = off; off += n; return o; };
    const size_t BHW = (size_t)NB * HW;
    size_t o_lab = alloc(2 * BHW);
    size_t o_l   = alloc(BHW);
    size_t o_pv[8], o_pi[8];
    o_pv[0] = o_l; o_pi[0] = 0;
    for (int l = 1; l < 8; ++l) o_pv[l] = alloc((size_t)NB * LHh[l] * LWw[l]);
    for (int l = 1; l < 8; ++l) o_pi[l] = alloc((size_t)NB * LHh[l] * LWw[l]);
    size_t o_iv = alloc(BHW), o_ii = alloc(BHW), o_ov = alloc(BHW), o_oi = alloc(BHW);
    size_t o_tot  = alloc(BHW);          // also reused for s and enh
    size_t o_tot8 = alloc(8 * BHW);
    size_t o_oall[7];                    // 16-channel conv outputs, levels 1..6
    for (int l = 1; l <= 6; ++l) o_oall[l] = alloc((size_t)NB * 16 * LHh[l] * LWw[l]);
    size_t o_fv = alloc(BHW), o_fi = alloc(BHW), o_fused = alloc(BHW);

    dim3 blk(256);
    dim3 gFull((HW + 255) / 256, 1, NB);
    unsigned* ppairs = stats + 16; // pairs 8..39

    k_init_pairs<<<1, 64, 0, stream>>>(stats, 64);
    k_rgb_lab<<<gFull, blk, 0, stream>>>(img_vis, Wf + o_l, Wf + o_lab);

    // Gaussian pyramids (vis L channel + IR)
    for (int l = 1; l < 8; ++l) {
        dim3 g((LHh[l] * LWw[l] + 255) / 256, 1, NB);
        k_pyr_down<<<g, blk, 0, stream>>>(Wf + o_pv[l - 1], Wf + o_pv[l],
                                          LHh[l - 1], LWw[l - 1], LHh[l], LWw[l]);
        const float* si = (l == 1) ? img_ir : (Wf + o_pi[l - 1]);
        k_pyr_down<<<g, blk, 0, stream>>>(si, Wf + o_pi[l],
                                          LHh[l - 1], LWw[l - 1], LHh[l], LWw[l]);
    }

    // intensity saliency (vis then ir)
    for (int srcI = 0; srcI < 2; ++srcI) {
        auto lvl = [&](int l) -> const float* {
            if (srcI == 0) return Wf + o_pv[l];
            return (l == 0) ? img_ir : (Wf + o_pi[l]);
        };
        float* intmap = Wf + (srcI == 0 ? o_iv : o_ii);
        k_zero<<<dim3(2048), blk, 0, stream>>>(Wf + o_tot, (long)BHW);
        for (int c = 1; c <= 3; ++c)
            for (int dd = 3; dd <= 4; ++dd) {
                int lo = c + dd;
                int hh = LHh[c], wh = LWw[c], hl = LHh[lo], wl = LWw[lo];
                k_init_pairs<<<1, 1, 0, stream>>>(stats, 1);
                dim3 gs((hh * wh + 255) / 256, 1, NB);
                k_int_stats<<<gs, blk, 0, stream>>>(lvl(c), lvl(lo), hh, wh, hl, wl, stats);
                k_int_accum<<<gFull, blk, 0, stream>>>(lvl(c), lvl(lo), hh, wh, hl, wl, c,
                                                       stats, Wf + o_tot);
            }
        k_init_pairs<<<1, 1, 0, stream>>>(stats, 1);
        k_minmax<<<dim3(512), blk, 0, stream>>>(Wf + o_tot, (long)BHW, stats);
        k_apply_mm<<<dim3((unsigned)((BHW + 255) / 256)), blk, 0, stream>>>(
            Wf + o_tot, intmap, (long)BHW, stats);
    }

    // orientation saliency (vis then ir) — both Gabor banks per level in one WMMA pass
    for (int srcI = 0; srcI < 2; ++srcI) {
        auto lvl = [&](int l) -> const float* {
            if (srcI == 0) return Wf + o_pv[l];
            return (l == 0) ? img_ir : (Wf + o_pi[l]);
        };
        float* orimap = Wf + (srcI == 0 ? o_ov : o_oi);
        k_zero<<<dim3(2048), blk, 0, stream>>>(Wf + o_tot8, (long)(8 * BHW));
        for (int l = 1; l <= 6; ++l) {
            dim3 gg((LWw[l] + 15) / 16, (LHh[l] + 7) / 8, NB);
            k_gabor16<<<gg, blk, 0, stream>>>(lvl(l), gabor, Wf + o_oall[l],
                                              LHh[l], LWw[l]);
        }
        for (int c = 1; c <= 3; ++c) {
            int hh = LHh[c], wh = LWw[c];
            for (int dd = 2; dd <= 3; ++dd) {
                int lo = c + dd;
                int hl = LHh[lo], wl = LWw[lo];
                int choff = (dd == 2) ? 0 : 8; // low conv uses bank0 (d=2) or bank1 (d=3)
                k_init_pairs<<<1, 32, 0, stream>>>(ppairs, 32);
                dim3 gs((hh * wh + 255) / 256, 1, NB * 8);
                k_ori_stats<<<gs, blk, 0, stream>>>(Wf + o_oall[c], Wf + o_oall[lo],
                                                    hh, wh, hl, wl, choff, ppairs);
                dim3 ga((HW + 255) / 256, 1, NB * 8);
                k_ori_accum<<<ga, blk, 0, stream>>>(Wf + o_oall[c], Wf + o_oall[lo],
                                                    hh, wh, hl, wl, c, choff, ppairs,
                                                    Wf + o_tot8);
            }
        }
        k_init_pairs<<<1, 32, 0, stream>>>(ppairs, 32);
        k_minmax_planes<<<dim3(64, 1, NB * 8), blk, 0, stream>>>(Wf + o_tot8, HW, ppairs);
        k_init_pairs<<<1, 1, 0, stream>>>(stats, 1);
        k_ori_s<<<gFull, blk, 0, stream>>>(Wf + o_tot8, ppairs, Wf + o_tot, stats);
        k_apply_mm<<<dim3((unsigned)((BHW + 255) / 256)), blk, 0, stream>>>(
            Wf + o_tot, orimap, (long)BHW, stats);
    }

    // fusion distances (deterministic block-per-batch reductions)
    k_dist<<<NB, blk, 0, stream>>>(Wf + o_iv, Wf + o_l, HW, dists + 0);
    k_dist<<<NB, blk, 0, stream>>>(Wf + o_ii, img_ir,   HW, dists + 4);
    k_dist<<<NB, blk, 0, stream>>>(Wf + o_ov, Wf + o_l, HW, dists + 8);
    k_dist<<<NB, blk, 0, stream>>>(Wf + o_oi, img_ir,   HW, dists + 12);

    // fusion maps
    k_init_pairs<<<1, 2, 0, stream>>>(stats + 2, 2); // pairs 1,2
    k_fusion<<<gFull, blk, 0, stream>>>(Wf + o_iv, Wf + o_ii, Wf + o_ov, Wf + o_oi,
                                        Wf + o_l, img_ir, dists,
                                        Wf + o_fv, Wf + o_fi, stats + 2, stats + 4);
    k_fuse<<<gFull, blk, 0, stream>>>(Wf + o_fv, Wf + o_fi, stats + 2, stats + 4,
                                      Wf + o_fused);

    // bilateral detail enhancement + Lab->RGB
    k_init_pairs<<<1, 1, 0, stream>>>(stats, 1);
    k_bilateral<<<gFull, blk, 0, stream>>>(Wf + o_fused, Wf + o_tot, stats);
    k_lab_out<<<gFull, blk, 0, stream>>>(Wf + o_tot, stats, Wf + o_lab, out);
}
